// TemporalCrossAttention_781684048573
// MI455X (gfx1250) — matrix operand order repaired and checked by reading, not compiled, not run on hardware
//
#include <hip/hip_runtime.h>

// ---------------------------------------------------------------------------
// TemporalCrossAttention for MI455X (gfx1250, wave32, WMMA bf16)
//   B=8 T=1024 D=1024 N=256 L=768 H=16 hd=64 TE=2048
// All big GEMMs + attention on v_wmma_f32_16x16x32_bf16; global->LDS staging
// uses CDNA5 GLOBAL_LOAD_ASYNC_TO_LDS_B128 (ASYNCcnt) where available.
// ---------------------------------------------------------------------------

#define B_ 8
#define T_ 1024
#define D_ 1024
#define N_ 256
#define L_ 768
#define H_ 16
#define HD_ 64
#define TE_ 2048
#define EPS_ 1e-5f

typedef __attribute__((ext_vector_type(16))) __bf16 v16bf;
typedef __attribute__((ext_vector_type(8)))  float  v8f;
typedef int v4i __attribute__((vector_size(16)));   // matches builtin param type

__device__ inline v8f wmma_bf16(v16bf a, v16bf b, v8f c) {
  return __builtin_amdgcn_wmma_f32_16x16x32_bf16(
      /*neg_a=*/false, a, /*neg_b=*/false, b,
      /*c_mod=*/(short)0, c, /*reuse_a=*/false, /*reuse_b=*/false);
}

// Load a 32-byte-contiguous fragment (16 bf16) from a 16B-aligned pointer.
__device__ inline v16bf ld_frag32(const __bf16* p) {
  union { uint4 u[2]; v16bf v; } f;
  f.u[0] = ((const uint4*)p)[0];
  f.u[1] = ((const uint4*)p)[1];
  return f.v;
}
// A-fragment: elements 0..15 contiguous at base+2*kh (kh in {0,8}).
__device__ inline v16bf ld_frag_a(const __bf16* base, int kh) {
  union { uint4 u[2]; v16bf v; } f;
  f.u[0] = *(const uint4*)(base + 2 * kh);
  f.u[1] = *(const uint4*)(base + 2 * kh + 8);
  return f.v;
}

// ---- CDNA5 async global->LDS (guarded: falls back to reg copies) ----------
#if __has_builtin(__builtin_amdgcn_global_load_async_to_lds_b128) && \
    __has_builtin(__builtin_amdgcn_s_wait_asynccnt)
#define ASYNC_LDS 1
__device__ inline void async_cp16(const void* g, void* l) {
  __builtin_amdgcn_global_load_async_to_lds_b128(
      (__attribute__((address_space(1))) v4i*)g,
      (__attribute__((address_space(3))) v4i*)l, 0, 0);
}
__device__ inline void async_wait0() { __builtin_amdgcn_s_wait_asynccnt(0); }
#else
#define ASYNC_LDS 0
__device__ inline void async_cp16(const void* g, void* l) {
  *(uint4*)l = *(const uint4*)g;
}
__device__ inline void async_wait0() {}
#endif

// ---------------------------------------------------------------------------
// float -> bf16 convert
// ---------------------------------------------------------------------------
__global__ __launch_bounds__(256) void f2bf_kernel(const float* __restrict__ s,
                                                   __bf16* __restrict__ d, int n) {
  int i = blockIdx.x * 256 + threadIdx.x;
  if (i < n) d[i] = (__bf16)s[i];
}

// ---------------------------------------------------------------------------
// Row LayerNorm (f32 in) -> bf16 out
// ---------------------------------------------------------------------------
__global__ __launch_bounds__(256) void ln_bf16_kernel(const float* __restrict__ x,
                                                      __bf16* __restrict__ out,
                                                      const float* __restrict__ g,
                                                      const float* __restrict__ bta,
                                                      int cols) {
  __shared__ float red[256];
  int row = blockIdx.x;
  const float* xr = x + (size_t)row * cols;
  float s = 0.f, s2 = 0.f;
  for (int c = threadIdx.x; c < cols; c += 256) { float v = xr[c]; s += v; s2 += v * v; }
  red[threadIdx.x] = s; __syncthreads();
  for (int st = 128; st > 0; st >>= 1) { if (threadIdx.x < st) red[threadIdx.x] += red[threadIdx.x + st]; __syncthreads(); }
  float mean = red[0] / cols; __syncthreads();
  red[threadIdx.x] = s2; __syncthreads();
  for (int st = 128; st > 0; st >>= 1) { if (threadIdx.x < st) red[threadIdx.x] += red[threadIdx.x + st]; __syncthreads(); }
  float var = red[0] / cols - mean * mean;
  float rstd = rsqrtf(var + EPS_);
  for (int c = threadIdx.x; c < cols; c += 256)
    out[(size_t)row * cols + c] = (__bf16)((xr[c] - mean) * rstd * g[c] + bta[c]);
}

// ---------------------------------------------------------------------------
// Generic bf16 WMMA GEMM: C[M,N] = (A[M,K] @ W[K,N] + bias (+resid)) * oscale
// Block tile 128x128, K-step 64, 256 threads = 8 waves (2x4 wave grid).
// 16 v_wmma per k-step per wave between each barrier pair.
// ---------------------------------------------------------------------------
#define TK_ 64
__global__ __launch_bounds__(256)
void gemm_bf16_kernel(const __bf16* __restrict__ A, const __bf16* __restrict__ W,
                      const float* __restrict__ bias, const float* __restrict__ resid,
                      float* __restrict__ outF, __bf16* __restrict__ outB,
                      int M, int N, int K, float oscale) {
  __shared__ __attribute__((aligned(16))) __bf16 As[128][TK_ + 8];   // A tile, row-major
  __shared__ __attribute__((aligned(16))) __bf16 Wt[128][TK_ + 8];   // W tile transposed [n][k]
  int tid = threadIdx.x;
  int wid = tid >> 5, lane = tid & 31;
  int wm = wid >> 2, wn = wid & 3;     // waves: 2 along M (64 rows), 4 along N (32 cols)
  int row0 = blockIdx.y * 128;
  int col0 = blockIdx.x * 128;
  int laneM = lane & 15, half = lane >> 4, kh = half * 8;

  v8f acc[4][2] = {};
  for (int k0 = 0; k0 < K; k0 += TK_) {
    __syncthreads();
    { // A tile: 128 rows x 64 cols, 2 threads per row, async b128 x4 each
      int r = tid >> 1, hh = (tid & 1) * 32;
      const __bf16* ga = A + (size_t)(row0 + r) * K + k0 + hh;
      __bf16* la = &As[r][hh];
      async_cp16(ga,      la);
      async_cp16(ga + 8,  la + 8);
      async_cp16(ga + 16, la + 16);
      async_cp16(ga + 24, la + 24);
      if (k0 + TK_ < K)
        __builtin_prefetch(ga + TK_, 0, 1);
    }
    { // W tile: 64 rows x 128 cols, register-transposed so B-frags are contiguous
      int wr = tid >> 2;            // 0..63 : k within tile
      int cg = (tid & 3) * 32;      // col group
      union { uint4 u[4]; __bf16 e[32]; } wv;
      const uint4* src = (const uint4*)(W + (size_t)(k0 + wr) * N + col0 + cg);
      wv.u[0] = src[0]; wv.u[1] = src[1]; wv.u[2] = src[2]; wv.u[3] = src[3];
#pragma unroll
      for (int j = 0; j < 32; ++j) Wt[cg + j][wr] = wv.e[j];
    }
    async_wait0();
    __syncthreads();

#pragma unroll
    for (int kk = 0; kk < 2; ++kk) {
      v16bf afrag[4], bfrag[2];
#pragma unroll
      for (int mt = 0; mt < 4; ++mt)
        afrag[mt] = ld_frag_a(&As[wm * 64 + mt * 16 + laneM][kk * 32], kh);
#pragma unroll
      for (int nt = 0; nt < 2; ++nt)
        bfrag[nt] = ld_frag32(&Wt[wn * 32 + nt * 16 + laneM][kk * 32 + half * 16]);
#pragma unroll
      for (int mt = 0; mt < 4; ++mt)
#pragma unroll
        for (int nt = 0; nt < 2; ++nt)
          acc[mt][nt] = wmma_bf16(afrag[mt], bfrag[nt], acc[mt][nt]);
    }
  }

  // Epilogue: C layout -> element i is row (i + 8*half), col laneM of its tile.
#pragma unroll
  for (int mt = 0; mt < 4; ++mt) {
#pragma unroll
    for (int nt = 0; nt < 2; ++nt) {
      int c = col0 + wn * 32 + nt * 16 + laneM;
      float bb = bias[c];
#pragma unroll
      for (int i = 0; i < 8; ++i) {
        int r = row0 + wm * 64 + mt * 16 + half * 8 + i;
        float v = acc[mt][nt][i] + bb;
        if (resid) v += resid[(size_t)r * N + c];
        v *= oscale;
        if (outF) outF[(size_t)r * N + c] = v;
        else      outB[(size_t)r * N + c] = (__bf16)v;
      }
    }
  }
}

// ---------------------------------------------------------------------------
// Fused cross-attention per (b,h): y = softmax(q k^T) v
// q pre-scaled by 1/sqrt(hd). K staged row-major in LDS (async), V transposed
// in LDS (register path). Flash-style online softmax over 32-token chunks;
// P converted C->A layout through a 1KB/wave LDS patch. 8 v_wmma per chunk.
// Grid: (B*H, T/128); 256 threads = 8 waves, 16 q-rows per wave.
// ---------------------------------------------------------------------------
#define KROW_ 72      // 64 + 8 pad (bf16 elems)
#define VROW_ 264     // 256 + 8 pad
__global__ __launch_bounds__(256)
void attn_kernel(const __bf16* __restrict__ q, const __bf16* __restrict__ k,
                 const __bf16* __restrict__ v, float* __restrict__ y) {
  extern __shared__ __align__(16) char smem[];
  __bf16* Ksm = (__bf16*)smem;            // [256][KROW_]
  __bf16* Vt  = Ksm + 256 * KROW_;        // [64][VROW_]  (V transposed: [d][n])
  __bf16* Pst = Vt + 64 * VROW_;          // [8 waves][16*32]

  int tid = threadIdx.x, wid = tid >> 5, lane = tid & 31;
  int b = blockIdx.x / H_, h = blockIdx.x % H_;
  int laneM = lane & 15, half = lane >> 4, kh = half * 8;

  { // stage K (row-major, async) and V (transposed, via regs); one token/thread
    int n = tid;  // 0..255
    const __bf16* ks = k + ((size_t)b * N_ + n) * D_ + h * HD_;
    __bf16* kd = &Ksm[n * KROW_];
#pragma unroll
    for (int j = 0; j < 8; ++j) async_cp16(ks + j * 8, kd + j * 8);
    union { uint4 u[8]; __bf16 e[64]; } vv;
    const uint4* vs = (const uint4*)(v + ((size_t)b * N_ + n) * D_ + h * HD_);
#pragma unroll
    for (int j = 0; j < 8; ++j) vv.u[j] = vs[j];
#pragma unroll
    for (int d = 0; d < HD_; ++d) Vt[d * VROW_ + n] = vv.e[d];
  }
  async_wait0();
  __syncthreads();

  int qr = blockIdx.y * 128 + wid * 16;

  // Q A-fragments for the two 32-wide d-chunks of hd=64
  const __bf16* qrow = q + ((size_t)b * T_ + qr + laneM) * D_ + h * HD_;
  v16bf aq0 = ld_frag_a(qrow, kh);
  v16bf aq1 = ld_frag_a(qrow + 32, kh);

  float mrow[8], lrow[8];
  v8f o[4] = {};
#pragma unroll
  for (int i = 0; i < 8; ++i) { mrow[i] = -3.0e38f; lrow[i] = 0.f; }
  __bf16* myP = Pst + wid * (16 * 32);

  for (int nc = 0; nc < N_ / 32; ++nc) {
    int n0 = nc * 32;
    // scores: S(16x32) = q(16x64) @ K^T(64x32); B-frag = K^T[d][n] = Ksm[n][d]
    v8f s0 = {}, s1 = {};
    s0 = wmma_bf16(aq0, ld_frag32(&Ksm[(n0 + laneM) * KROW_ + half * 16]), s0);
    s0 = wmma_bf16(aq1, ld_frag32(&Ksm[(n0 + laneM) * KROW_ + 32 + half * 16]), s0);
    s1 = wmma_bf16(aq0, ld_frag32(&Ksm[(n0 + 16 + laneM) * KROW_ + half * 16]), s1);
    s1 = wmma_bf16(aq1, ld_frag32(&Ksm[(n0 + 16 + laneM) * KROW_ + 32 + half * 16]), s1);

    // online softmax; element i <-> row (half*8 + i), col laneM (+16 for s1)
#pragma unroll
    for (int i = 0; i < 8; ++i) {
      float t = fmaxf(s0[i], s1[i]);
#pragma unroll
      for (int m2 = 1; m2 < 16; m2 <<= 1) t = fmaxf(t, __shfl_xor(t, m2, 16));
      float mn = fmaxf(mrow[i], t);
      float corr = __expf(mrow[i] - mn);
      mrow[i] = mn;
      lrow[i] *= corr;
      o[0][i] *= corr; o[1][i] *= corr; o[2][i] *= corr; o[3][i] *= corr;
      float p0 = __expf(s0[i] - mn);
      float p1 = __expf(s1[i] - mn);
      float rs = p0 + p1;
#pragma unroll
      for (int m2 = 1; m2 < 16; m2 <<= 1) rs += __shfl_xor(rs, m2, 16);
      lrow[i] += rs;
      int r = half * 8 + i;
      myP[r * 32 + laneM]      = (__bf16)p0;
      myP[r * 32 + 16 + laneM] = (__bf16)p1;
    }
    // P(16x32) @ V(32x64): A-frag of P from LDS patch (wave-private; DS in-order)
    v16bf pf = ld_frag_a(myP + laneM * 32, kh);
#pragma unroll
    for (int t2 = 0; t2 < 4; ++t2) {
      v16bf vf = ld_frag32(&Vt[(t2 * 16 + laneM) * VROW_ + n0 + half * 16]);
      o[t2] = wmma_bf16(pf, vf, o[t2]);
    }
  }

  // normalize and write y (f32, (B,T,D))
  float rl[8];
#pragma unroll
  for (int i = 0; i < 8; ++i) rl[i] = 1.0f / lrow[i];
#pragma unroll
  for (int t2 = 0; t2 < 4; ++t2)
#pragma unroll
    for (int i = 0; i < 8; ++i) {
      int r = qr + half * 8 + i;
      y[((size_t)b * T_ + r) * D_ + h * HD_ + t2 * 16 + laneM] = o[t2][i] * rl[i];
    }
}

// ---------------------------------------------------------------------------
// emb_out = silu(emb) @ Wemb + bemb   (tiny: 8 x 2048 <- 8 x 2048 @ 2048 x 2048)
// ---------------------------------------------------------------------------
__global__ __launch_bounds__(128)
void emb_gemm_kernel(const float* __restrict__ emb, const float* __restrict__ Wemb,
                     const float* __restrict__ bemb, float* __restrict__ eo) {
  int col = blockIdx.x * 128 + threadIdx.x;
  int b = blockIdx.y;
  const float* e = emb + (size_t)b * TE_;
  float acc = 0.f;
  for (int i = 0; i < TE_; ++i) {
    float ev = e[i];
    float s = ev / (1.f + __expf(-ev));
    acc += s * Wemb[(size_t)i * (2 * D_) + col];
  }
  eo[(size_t)b * (2 * D_) + col] = acc + bemb[col];
}

// ---------------------------------------------------------------------------
// h = silu( LN(y)*(1+scale) + shift ) -> bf16 (ready for the out-proj GEMM)
// ---------------------------------------------------------------------------
__global__ __launch_bounds__(256)
void stylize_kernel(const float* __restrict__ y, const float* __restrict__ eo,
                    const float* __restrict__ g, const float* __restrict__ bta,
                    __bf16* __restrict__ out) {
  __shared__ float red[256];
  int row = blockIdx.x;
  int b = row / T_;
  const float* yr = y + (size_t)row * D_;
  const float* sc = eo + (size_t)b * (2 * D_);  // [0..D) scale, [D..2D) shift
  float s = 0.f, s2 = 0.f;
  for (int c = threadIdx.x; c < D_; c += 256) { float v = yr[c]; s += v; s2 += v * v; }
  red[threadIdx.x] = s; __syncthreads();
  for (int st = 128; st > 0; st >>= 1) { if (threadIdx.x < st) red[threadIdx.x] += red[threadIdx.x + st]; __syncthreads(); }
  float mean = red[0] / D_; __syncthreads();
  red[threadIdx.x] = s2; __syncthreads();
  for (int st = 128; st > 0; st >>= 1) { if (threadIdx.x < st) red[threadIdx.x] += red[threadIdx.x + st]; __syncthreads(); }
  float rstd = rsqrtf(red[0] / D_ - mean * mean + EPS_);
  for (int c = threadIdx.x; c < D_; c += 256) {
    float v = (yr[c] - mean) * rstd * g[c] + bta[c];
    v = v * (1.f + sc[c]) + sc[D_ + c];
    v = v / (1.f + __expf(-v));  // silu
    out[(size_t)row * D_ + c] = (__bf16)v;
  }
}

// ---------------------------------------------------------------------------
extern "C" void kernel_launch(void* const* d_in, const int* in_sizes, int n_in,
                              void* d_out, int out_size, void* d_ws, size_t ws_size,
                              hipStream_t stream) {
  (void)in_sizes; (void)n_in; (void)out_size; (void)ws_size;
  const float* x     = (const float*)d_in[0];
  const float* xf    = (const float*)d_in[1];
  const float* emb   = (const float*)d_in[2];
  const float* ln_g  = (const float*)d_in[3];
  const float* ln_b  = (const float*)d_in[4];
  const float* cln_g = (const float*)d_in[5];
  const float* cln_b = (const float*)d_in[6];
  const float* Wq    = (const float*)d_in[7];
  const float* bq    = (const float*)d_in[8];
  const float* Wk    = (const float*)d_in[9];
  const float* bk    = (const float*)d_in[10];
  const float* Wv    = (const float*)d_in[11];
  const float* bv    = (const float*)d_in[12];
  const float* sln_g = (const float*)d_in[13];
  const float* sln_b = (const float*)d_in[14];
  const float* Wemb  = (const float*)d_in[15];
  const float* bemb  = (const float*)d_in[16];
  const float* Wout  = (const float*)d_in[17];
  const float* bout  = (const float*)d_in[18];
  float* out = (float*)d_out;

  // workspace carve-out (256B aligned)
  char* cur = (char*)d_ws;
  auto alloc = [&](size_t bytes) -> char* {
    char* p = cur; cur += (bytes + 255) & ~(size_t)255; return p;
  };
  __bf16* xln   = (__bf16*)alloc((size_t)B_ * T_ * D_ * 2);     // 16 MB
  __bf16* xfn   = (__bf16*)alloc((size_t)B_ * N_ * L_ * 2);     //  3 MB
  __bf16* Wqb   = (__bf16*)alloc((size_t)D_ * D_ * 2);          //  2 MB
  __bf16* Wkb   = (__bf16*)alloc((size_t)L_ * D_ * 2);          // 1.5 MB
  __bf16* Wvb   = (__bf16*)alloc((size_t)L_ * D_ * 2);          // 1.5 MB
  __bf16* Woutb = (__bf16*)alloc((size_t)D_ * D_ * 2);          //  2 MB
  __bf16* qb    = (__bf16*)alloc((size_t)B_ * T_ * D_ * 2);     // 16 MB
  __bf16* kb    = (__bf16*)alloc((size_t)B_ * N_ * D_ * 2);     //  4 MB
  __bf16* vb    = (__bf16*)alloc((size_t)B_ * N_ * D_ * 2);     //  4 MB
  float*  yb    = (float*)alloc((size_t)B_ * T_ * D_ * 4);      // 32 MB
  float*  eo    = (float*)alloc((size_t)B_ * 2 * D_ * 4);       // 64 KB
  __bf16* hb    = (__bf16*)alloc((size_t)B_ * T_ * D_ * 2);     // 16 MB

  // 1) weights -> bf16
  f2bf_kernel<<<(D_ * D_ + 255) / 256, 256, 0, stream>>>(Wq, Wqb, D_ * D_);
  f2bf_kernel<<<(L_ * D_ + 255) / 256, 256, 0, stream>>>(Wk, Wkb, L_ * D_);
  f2bf_kernel<<<(L_ * D_ + 255) / 256, 256, 0, stream>>>(Wv, Wvb, L_ * D_);
  f2bf_kernel<<<(D_ * D_ + 255) / 256, 256, 0, stream>>>(Wout, Woutb, D_ * D_);

  // 2) LayerNorms -> bf16 activations
  ln_bf16_kernel<<<B_ * T_, 256, 0, stream>>>(x, xln, ln_g, ln_b, D_);
  ln_bf16_kernel<<<B_ * N_, 256, 0, stream>>>(xf, xfn, cln_g, cln_b, L_);

  // 3) projections (q scaled by 1/sqrt(hd)=0.125)
  gemm_bf16_kernel<<<dim3(D_ / 128, (B_ * T_) / 128), 256, 0, stream>>>(
      xln, Wqb, bq, nullptr, nullptr, qb, B_ * T_, D_, D_, 0.125f);
  gemm_bf16_kernel<<<dim3(D_ / 128, (B_ * N_) / 128), 256, 0, stream>>>(
      xfn, Wkb, bk, nullptr, nullptr, kb, B_ * N_, D_, L_, 1.0f);
  gemm_bf16_kernel<<<dim3(D_ / 128, (B_ * N_) / 128), 256, 0, stream>>>(
      xfn, Wvb, bv, nullptr, nullptr, vb, B_ * N_, D_, L_, 1.0f);

  // 4) fused attention (dynamic LDS: K 36864B + Vt 33792B + P 8192B)
  size_t attn_lds = (256 * KROW_ + 64 * VROW_ + 8 * 16 * 32) * sizeof(__bf16);
  attn_kernel<<<dim3(B_ * H_, T_ / 128), 256, attn_lds, stream>>>(qb, kb, vb, yb);

  // 5) stylization
  emb_gemm_kernel<<<dim3((2 * D_) / 128, B_), 128, 0, stream>>>(emb, Wemb, bemb, eo);
  stylize_kernel<<<B_ * T_, 256, 0, stream>>>(yb, eo, sln_g, sln_b, hb);

  // 6) out = x + h @ Wout + bout
  gemm_bf16_kernel<<<dim3(D_ / 128, (B_ * T_) / 128), 256, 0, stream>>>(
      hb, Woutb, bout, x, out, nullptr, B_ * T_, D_, D_, 1.0f);
}
